// block_27650999451989
// MI455X (gfx1250) — compile-verified, matrix operand order checked by reading, not run on hardware
//
#include <hip/hip_runtime.h>

typedef __attribute__((ext_vector_type(16))) _Float16 v16h;
typedef __attribute__((ext_vector_type(8)))  float    v8f;
typedef __attribute__((ext_vector_type(4)))  float    f32x4;

#define MTOK   65536          // B*T tokens
#define CDIM   384
#define TDIM   128
#define HDIM   16
#define DDIM   24
#define DPAD   32
#define FFDIM  1536
#define BHEADS 8192           // B*H
#define HEADELEMS ((size_t)BHEADS * TDIM * DPAD)   // elems per k/q/v buffer

// ---------------------------------------------------------------------------
// WMMA fragment loader. A-matrix layout per CDNA5 ISA 7.12.2 (16-bit, 16x32):
//   lanes 0-15  : row = row0+lane,    halves[0..7]=K k0..k0+7, [8..15]=k0+16..k0+23
//   lanes 16-31 : row = row0+lane-16, halves[0..7]=k0+8..k0+15, [8..15]=k0+24..k0+31
// B-matrix assumed N-mirrored (lane indexes column), so loading W row-major
// [N,K] with row0 = n0 yields D = A * W^T, which is what every layer needs.
// Works for global and LDS pointers (rows must be 16B aligned).
// ---------------------------------------------------------------------------
__device__ __forceinline__ v16h load_frag16(const _Float16* base, int ld, int row0, int k0) {
  const int lane = threadIdx.x & 31;
  const int hi   = lane >> 4;
  const int r    = row0 + (lane & 15);
  const _Float16* p = base + (size_t)r * ld + k0 + hi * 8;
  v16h f;
  *reinterpret_cast<f32x4*>(&f)       = *reinterpret_cast<const f32x4*>(p);
  *(reinterpret_cast<f32x4*>(&f) + 1) = *reinterpret_cast<const f32x4*>(p + 16);
  return f;
}

// ---------------------------------------------------------------------------
// Per-wave 16 x (16*NT) output strip: one shared A-fragment per K-step feeds
// NT back-to-back WMMAs. All NT B-fragments are loaded before the first WMMA
// so the scheduler can issue staggered s_wait_loadcnt (later loads overlap
// earlier WMMAs) instead of a full wait per matrix op.
// ---------------------------------------------------------------------------
template <int NT>
__device__ __forceinline__ void wmma_strip(const _Float16* A, int lda,
                                           const _Float16* W, int ldw,
                                           int m0, int n0, int K, v8f acc[NT]) {
#pragma unroll
  for (int t = 0; t < NT; ++t) acc[t] = {};
#pragma unroll 2
  for (int k0 = 0; k0 < K; k0 += 32) {
    if (k0 + 32 < K) {  // lowers to global_prefetch_b8 on gfx1250
      __builtin_prefetch(A + (size_t)m0 * lda + k0 + 32, 0, 1);
      __builtin_prefetch(W + (size_t)n0 * ldw + k0 + 32, 0, 1);
    }
    v16h a = load_frag16(A, lda, m0, k0);
    v16h b[NT];
#pragma unroll
    for (int t = 0; t < NT; ++t)
      b[t] = load_frag16(W, ldw, n0 + 16 * t, k0);
#pragma unroll
    for (int t = 0; t < NT; ++t)
      acc[t] = __builtin_amdgcn_wmma_f32_16x16x32_f16(false, a, false, b[t],
                                                      (short)0, acc[t], false, false);
  }
}

// ---------------------------------------------------------------------------
// fp32 -> fp16 weight conversion (grid-stride)
// ---------------------------------------------------------------------------
__global__ void cvt_kernel(const float* __restrict__ src, _Float16* __restrict__ dst, int n) {
  for (int i = blockIdx.x * blockDim.x + threadIdx.x; i < n; i += gridDim.x * blockDim.x)
    dst[i] = (_Float16)src[i];
}

// ---------------------------------------------------------------------------
// LayerNorm over C=384. One row per 128-thread block (3 elems/thread).
// outf may alias x (in-place) or be null.
// ---------------------------------------------------------------------------
__global__ __launch_bounds__(128) void ln_kernel(const float* x,
                                                 const float* __restrict__ g,
                                                 const float* __restrict__ b,
                                                 _Float16* __restrict__ outh,
                                                 float* outf) {
  const size_t row = blockIdx.x;
  const float* xr = x + row * CDIM;
  const int tid = threadIdx.x;
  float v0 = xr[tid], v1 = xr[tid + 128], v2 = xr[tid + 256];
  float s  = v0 + v1 + v2;
  float s2 = v0 * v0 + v1 * v1 + v2 * v2;
#pragma unroll
  for (int off = 16; off; off >>= 1) {
    s  += __shfl_xor(s,  off, 32);
    s2 += __shfl_xor(s2, off, 32);
  }
  __shared__ float rs[4], rq[4];
  const int wid = tid >> 5, lane = tid & 31;
  if (!lane) { rs[wid] = s; rq[wid] = s2; }
  __syncthreads();
  const float tot  = rs[0] + rs[1] + rs[2] + rs[3];
  const float tot2 = rq[0] + rq[1] + rq[2] + rq[3];
  const float mu   = tot * (1.0f / CDIM);
  const float var  = tot2 * (1.0f / CDIM) - mu * mu;
  const float rstd = rsqrtf(var + 1e-5f);
  const int c0 = tid, c1 = tid + 128, c2 = tid + 256;
  float y0 = (v0 - mu) * rstd * g[c0] + b[c0];
  float y1 = (v1 - mu) * rstd * g[c1] + b[c1];
  float y2 = (v2 - mu) * rstd * g[c2] + b[c2];
  outh[row * CDIM + c0] = (_Float16)y0;
  outh[row * CDIM + c1] = (_Float16)y1;
  outh[row * CDIM + c2] = (_Float16)y2;
  if (outf) {
    outf[row * CDIM + c0] = y0;
    outf[row * CDIM + c1] = y1;
    outf[row * CDIM + c2] = y2;
  }
}

// ---------------------------------------------------------------------------
// QKV GEMM: y1h[M,384] x Wqkv[1152,384]^T, scattered to [3][B,H,T,32] f16.
// Wqkv rows: 0..383 = wk, 384..767 = wq, 768..1151 = wv (f = h*24+d order).
// 2 waves/block, each wave owns a 16x64 strip -> grid (M/16, 1152/128).
// ---------------------------------------------------------------------------
__global__ __launch_bounds__(64) void qkv_kernel(const _Float16* __restrict__ A,
                                                 const _Float16* __restrict__ W,
                                                 _Float16* __restrict__ kqv) {
  const int w  = threadIdx.x >> 5;
  const int m0 = blockIdx.x * 16;
  const int n0 = (blockIdx.y * 2 + w) * 64;
  v8f acc[4];
  wmma_strip<4>(A, CDIM, W, CDIM, m0, n0, CDIM, acc);
  const int lane = threadIdx.x & 31;
  const int nloc = lane & 15, hi = lane >> 4;
#pragma unroll
  for (int t = 0; t < 4; ++t) {
    const int n   = n0 + 16 * t + nloc;
    const int mat = n / CDIM;
    const int fi  = n % CDIM;
    const int h   = fi / DDIM;
    const int d   = fi % DDIM;
    _Float16* dst = kqv + (size_t)mat * HEADELEMS;
#pragma unroll
    for (int gg = 0; gg < 8; ++gg) {
      const int m = m0 + gg + hi * 8;
      const int bb = m >> 7, tt = m & 127;
      dst[(((size_t)(bb * HDIM + h) * TDIM) + tt) * DPAD + d] = (_Float16)acc[t][gg];
    }
  }
}

// ---------------------------------------------------------------------------
// Attention: one block per (b,h). 8 waves, wave w owns query rows [16w,16w+16).
// scores = K @ Q^T (reference swaps q/k, no 1/sqrt(d) scale), causal mask,
// softmax, att = P @ V. Output written in head-concat layout -> attc[M,384].
// ---------------------------------------------------------------------------
__global__ __launch_bounds__(256) void attn_kernel(const _Float16* __restrict__ kq,
                                                   const _Float16* __restrict__ qq,
                                                   const _Float16* __restrict__ vq,
                                                   _Float16* __restrict__ attc) {
  __shared__ __align__(16) _Float16 VT[DPAD][136];     // V transposed [d][s]
  __shared__ __align__(16) _Float16 P[8][16][136];     // per-wave probs [t][s]
  const int bh = blockIdx.x;
  const int b  = bh >> 4, h = bh & 15;
  const size_t base = (size_t)bh * (TDIM * DPAD);
  const int tid = threadIdx.x;

  // stage V transposed (pads d=24..31 are zero in vq -> stay zero)
  for (int idx = tid; idx < TDIM * DPAD; idx += 256) {
    const int s = idx >> 5, d = idx & 31;
    VT[d][s] = vq[base + idx];
  }
  __syncthreads();

  const int w    = tid >> 5;
  const int lane = tid & 31;
  const int nloc = lane & 15, hi = lane >> 4;
  const int m0   = w * 16;

  // scores: A = K rows (fixed), B = Q rows as columns, K-dim = 32 (padded)
  const v16h ak = load_frag16(kq + base, DPAD, m0, 0);
  v8f sacc[8];
  v8f zero = {};
#pragma unroll
  for (int j = 0; j < 8; ++j) {
    v16h bq = load_frag16(qq + base, DPAD, j * 16, 0);
    sacc[j] = __builtin_amdgcn_wmma_f32_16x16x32_f16(false, ak, false, bq,
                                                     (short)0, zero, false, false);
  }

  // causal mask + row softmax (row t lives at fixed (g,half); cols span
  // 8 tiles x 16 lanes of the half-wave)
  float rmax[8];
#pragma unroll
  for (int gg = 0; gg < 8; ++gg) rmax[gg] = -3.0e38f;
#pragma unroll
  for (int j = 0; j < 8; ++j) {
    const int sg = j * 16 + nloc;
#pragma unroll
    for (int gg = 0; gg < 8; ++gg) {
      const int tg = m0 + gg + hi * 8;
      float val = (sg <= tg) ? sacc[j][gg] : -3.0e38f;
      sacc[j][gg] = val;
      rmax[gg] = fmaxf(rmax[gg], val);
    }
  }
#pragma unroll
  for (int gg = 0; gg < 8; ++gg)
#pragma unroll
    for (int off = 8; off; off >>= 1)
      rmax[gg] = fmaxf(rmax[gg], __shfl_xor(rmax[gg], off, 16));
  float rsum[8] = {0.f, 0.f, 0.f, 0.f, 0.f, 0.f, 0.f, 0.f};
#pragma unroll
  for (int j = 0; j < 8; ++j)
#pragma unroll
    for (int gg = 0; gg < 8; ++gg) {
      const float e = __expf(sacc[j][gg] - rmax[gg]);
      sacc[j][gg] = e;
      rsum[gg] += e;
    }
#pragma unroll
  for (int gg = 0; gg < 8; ++gg) {
#pragma unroll
    for (int off = 8; off; off >>= 1) rsum[gg] += __shfl_xor(rsum[gg], off, 16);
    rsum[gg] = 1.0f / rsum[gg];
  }

  // stash probabilities into this wave's LDS region as f16
#pragma unroll
  for (int j = 0; j < 8; ++j) {
    const int col = j * 16 + nloc;
#pragma unroll
    for (int gg = 0; gg < 8; ++gg)
      P[w][gg + hi * 8][col] = (_Float16)(sacc[j][gg] * rsum[gg]);
  }
  __syncthreads();

  // att = P[16,128] @ V[128,32]; two 16-col output tiles (d 0..15, 16..31)
  v8f oacc0 = {}, oacc1 = {};
#pragma unroll
  for (int kc = 0; kc < TDIM; kc += 32) {
    v16h pa  = load_frag16(&P[w][0][0], 136, 0, kc);
    v16h vb0 = load_frag16(&VT[0][0], 136, 0, kc);
    v16h vb1 = load_frag16(&VT[0][0], 136, 16, kc);
    oacc0 = __builtin_amdgcn_wmma_f32_16x16x32_f16(false, pa, false, vb0,
                                                   (short)0, oacc0, false, false);
    oacc1 = __builtin_amdgcn_wmma_f32_16x16x32_f16(false, pa, false, vb1,
                                                   (short)0, oacc1, false, false);
  }

  // write valid d<24 in concat layout: attc[(b*T+t)*384 + h*24 + d]
#pragma unroll
  for (int gg = 0; gg < 8; ++gg) {
    const int t = m0 + gg + hi * 8;
    const size_t rowoff = ((size_t)(b * TDIM + t)) * CDIM + h * DDIM;
    attc[rowoff + nloc] = (_Float16)oacc0[gg];
    if (nloc < 8) attc[rowoff + 16 + nloc] = (_Float16)oacc1[gg];
  }
}

// ---------------------------------------------------------------------------
// GEMM + bias + f32 residual -> f32 out  (used for attn-proj and FFN2)
// 2 waves/block, 16x64 strip per wave.
// ---------------------------------------------------------------------------
__global__ __launch_bounds__(64) void gemm_bias_res_kernel(const _Float16* __restrict__ A,
                                                           const _Float16* __restrict__ W,
                                                           const float* __restrict__ bias,
                                                           const float* __restrict__ res,
                                                           float* __restrict__ out,
                                                           int K, int N) {
  const int w  = threadIdx.x >> 5;
  const int m0 = blockIdx.x * 16;
  const int n0 = (blockIdx.y * 2 + w) * 64;
  v8f acc[4];
  wmma_strip<4>(A, K, W, K, m0, n0, K, acc);
  const int lane = threadIdx.x & 31;
  const int nloc = lane & 15, hi = lane >> 4;
#pragma unroll
  for (int t = 0; t < 4; ++t) {
    const int n = n0 + 16 * t + nloc;
    const float bn = bias[n];
#pragma unroll
    for (int gg = 0; gg < 8; ++gg) {
      const size_t m = m0 + gg + hi * 8;
      out[m * N + n] = acc[t][gg] + bn + res[m * N + n];
    }
  }
}

// ---------------------------------------------------------------------------
// FFN1: GEMM + bias + exact GELU -> f16 out [M,1536]; 16x64 strip per wave.
// ---------------------------------------------------------------------------
__global__ __launch_bounds__(64) void ffn1_kernel(const _Float16* __restrict__ A,
                                                  const _Float16* __restrict__ W,
                                                  const float* __restrict__ bias,
                                                  _Float16* __restrict__ out) {
  const int w  = threadIdx.x >> 5;
  const int m0 = blockIdx.x * 16;
  const int n0 = (blockIdx.y * 2 + w) * 64;
  v8f acc[4];
  wmma_strip<4>(A, CDIM, W, CDIM, m0, n0, CDIM, acc);
  const int lane = threadIdx.x & 31;
  const int nloc = lane & 15, hi = lane >> 4;
#pragma unroll
  for (int t = 0; t < 4; ++t) {
    const int n = n0 + 16 * t + nloc;
    const float bn = bias[n];
#pragma unroll
    for (int gg = 0; gg < 8; ++gg) {
      const size_t m = m0 + gg + hi * 8;
      const float v = acc[t][gg] + bn;
      const float gl = 0.5f * v * (1.0f + erff(v * 0.70710678118654752f));
      out[m * FFDIM + n] = (_Float16)gl;
    }
  }
}

// ---------------------------------------------------------------------------
// Host-side orchestration
// ---------------------------------------------------------------------------
extern "C" void kernel_launch(void* const* d_in, const int* in_sizes, int n_in,
                              void* d_out, int out_size, void* d_ws, size_t ws_size,
                              hipStream_t stream) {
  (void)in_sizes; (void)n_in; (void)out_size; (void)ws_size;
  const float* x     = (const float*)d_in[0];
  const float* wk    = (const float*)d_in[1];
  const float* wq    = (const float*)d_in[2];
  const float* wv    = (const float*)d_in[3];
  const float* projw = (const float*)d_in[4];
  const float* projb = (const float*)d_in[5];
  const float* ln1g  = (const float*)d_in[6];
  const float* ln1b  = (const float*)d_in[7];
  const float* ln2g  = (const float*)d_in[8];
  const float* ln2b  = (const float*)d_in[9];
  const float* w1    = (const float*)d_in[10];
  const float* b1    = (const float*)d_in[11];
  const float* w2    = (const float*)d_in[12];
  const float* b2    = (const float*)d_in[13];
  float* out = (float*)d_out;

  char* ws = (char*)d_ws;
  size_t off = 0;
  auto take = [&](size_t bytes) -> char* {
    char* p = ws + off;
    off += (bytes + 255) & ~(size_t)255;
    return p;
  };
  _Float16* wqkv_h  = (_Float16*)take((size_t)1152 * 384 * 2);
  _Float16* projw_h = (_Float16*)take((size_t)384 * 384 * 2);
  _Float16* w1_h    = (_Float16*)take((size_t)1536 * 384 * 2);
  _Float16* w2_h    = (_Float16*)take((size_t)384 * 1536 * 2);
  // y1h reused later as attc, then as xlnh (lifetimes disjoint, stream-ordered)
  _Float16* y1h     = (_Float16*)take((size_t)MTOK * CDIM * 2);
  // kqv (3 head buffers) reused later as FFN hidden hh (exact same size)
  _Float16* kqv     = (_Float16*)take(3 * HEADELEMS * 2);
  // x2 reused in-place as post-LN2 xln
  float*    x2      = (float*)take((size_t)MTOK * CDIM * 4);

  // 1. weights -> f16 (wk/wq/wv are [H,D,C] flat == [384,384] in f=h*24+d order)
  cvt_kernel<<<dim3(512), 256, 0, stream>>>(wk, wqkv_h, 147456);
  cvt_kernel<<<dim3(512), 256, 0, stream>>>(wq, wqkv_h + 147456, 147456);
  cvt_kernel<<<dim3(512), 256, 0, stream>>>(wv, wqkv_h + 294912, 147456);
  cvt_kernel<<<dim3(512), 256, 0, stream>>>(projw, projw_h, 147456);
  cvt_kernel<<<dim3(1024), 256, 0, stream>>>(w1, w1_h, 589824);
  cvt_kernel<<<dim3(1024), 256, 0, stream>>>(w2, w2_h, 589824);

  // 2. zero k/q/v pads (d = 24..31) so the 32-wide WMMA K-dim is exact
  hipMemsetAsync(kqv, 0, 3 * HEADELEMS * 2, stream);

  // 3. LN1 -> f16
  ln_kernel<<<dim3(MTOK), 128, 0, stream>>>(x, ln1g, ln1b, y1h, nullptr);

  // 4. QKV GEMM -> [3][B,H,T,32]  (N=1152 = 9 * 128-wide blocks)
  qkv_kernel<<<dim3(4096, 9), 64, 0, stream>>>(y1h, wqkv_h, kqv);

  // 5. attention -> attc (reuses y1h region)
  _Float16* kb = kqv;
  _Float16* qb = kqv + HEADELEMS;
  _Float16* vb = kqv + 2 * HEADELEMS;
  attn_kernel<<<dim3(BHEADS), 256, 0, stream>>>(kb, qb, vb, y1h);

  // 6. attn proj + bias + residual(x) -> x2 (f32)   (N=384 = 3 * 128)
  gemm_bias_res_kernel<<<dim3(4096, 3), 64, 0, stream>>>(y1h, projw_h, projb, x,
                                                         x2, CDIM, CDIM);

  // 7. LN2: x2 -> xln (f32, in place) + xlnh (f16, reuses y1h region)
  ln_kernel<<<dim3(MTOK), 128, 0, stream>>>(x2, ln2g, ln2b, y1h, x2);

  // 8. FFN1 + GELU -> hh f16 (reuses kqv region)   (N=1536 = 12 * 128)
  ffn1_kernel<<<dim3(4096, 12), 64, 0, stream>>>(y1h, w1_h, b1, kqv);

  // 9. FFN2 + bias + residual(xln) -> out f32      (N=384 = 3 * 128)
  gemm_bias_res_kernel<<<dim3(4096, 3), 64, 0, stream>>>(kqv, w2_h, b2, x2,
                                                         out, FFDIM, CDIM);
}
